// MultiHeadAttentionAggregator_90288802497385
// MI455X (gfx1250) — matrix operand order repaired and checked by reading, not compiled, not run on hardware
//
#include <hip/hip_runtime.h>

#define HH     8
#define LROWS  4096
#define MROWS  1024
#define PROWS  4096
#define HIDDEN 512
#define DHEAD  64
#define DOUT   256
#define AWCOLS (HH * MROWS)   // 8192

typedef __attribute__((ext_vector_type(16))) __bf16       v16bf;
typedef __attribute__((ext_vector_type(8)))  unsigned int v8ui;
typedef __attribute__((ext_vector_type(8)))  float        v8f;

__device__ __forceinline__ unsigned short f32_to_bf16(float f) {
  unsigned int u = __float_as_uint(f);
  u += 0x7FFFu + ((u >> 16) & 1u);          // round-to-nearest-even
  return (unsigned short)(u >> 16);
}

// Build a 16-element bf16 fragment from two 16B loads at p and p+gap (ushorts).
__device__ __forceinline__ v16bf frag16(const unsigned short* __restrict__ p, int gap) {
  uint4 x0 = *(const uint4*)(p);
  uint4 x1 = *(const uint4*)(p + gap);
  v8ui w;
  w[0] = x0.x; w[1] = x0.y; w[2] = x0.z; w[3] = x0.w;
  w[4] = x1.x; w[5] = x1.y; w[6] = x1.z; w[7] = x1.w;
  return __builtin_bit_cast(v16bf, w);
}

// ---------------------------------------------------------------------------
// Batched bf16 GEMM, f32 accumulate.  C = alpha*A·B (+bias) (+addsrc) (relu?)
//   A  : [M,K] bf16 row-major (K contiguous)
//   BT : [N,K] bf16 row-major (i.e. B transposed; K contiguous)
// Outputs (any subset): Cf f32 [M,N]; Cb bf16 [M,N]; CbT bf16 [N,M].
// Requirements: M % 128 == 0, N % 64 == 0, K % 32 == 0.
// 256 threads = 8 waves; block tile 128x128; wave tile 32x64 (2x4 WMMAs).
// Inner loop is 12 global_load_b128 + 8 v_wmma_f32_16x16x32_bf16.
// ---------------------------------------------------------------------------
__global__ __launch_bounds__(256)
void gemm_bf16_kernel(const unsigned short* __restrict__ A, long ldA, long zsA,
                      const unsigned short* __restrict__ BT, long ldBT, long zsB,
                      float* __restrict__ Cf, long ldc, long zsC,
                      unsigned short* __restrict__ Cb, long ldcb, long zsCb,
                      unsigned short* __restrict__ CbT, long ldcT, long zsCT,
                      const float* __restrict__ bias,
                      const float* __restrict__ addsrc, long ldadd,
                      int Ndim, int Kdim, float alpha, int relu) {
  const int z = blockIdx.z;
  A  += (long)z * zsA;
  BT += (long)z * zsB;

  const int lane    = threadIdx.x & 31;
  const int wv      = threadIdx.x >> 5;
  const int rowBase = blockIdx.y * 128 + (wv >> 1) * 32;
  const int colBase = blockIdx.x * 128 + (wv & 1) * 64;
  if (colBase >= Ndim) return;           // wave-uniform; EXEC stays all-ones

  // A frag: lane&15 = row; lanes 0-15 hold K 0-15,
  //         lanes 16-31 hold K 16-31  -> gap 8.
  const unsigned short* pa0 = A + (long)(rowBase + (lane & 15)) * ldA + ((lane >> 4) << 4);
  const unsigned short* pa1 = pa0 + ((long)ldA << 4);
  // B frag: lane&15 = n; lanes 0-15 hold K 0-15, lanes 16-31 K 16-31 -> gap 8.
  const unsigned short* pb0 = BT + (long)(colBase + (lane & 15)) * ldBT + ((lane >> 4) << 4);
  const unsigned short* pb1 = pb0 + ((long)ldBT << 4);
  const unsigned short* pb2 = pb1 + ((long)ldBT << 4);
  const unsigned short* pb3 = pb2 + ((long)ldBT << 4);

  v8f acc[2][4];
#pragma unroll
  for (int i = 0; i < 2; ++i)
#pragma unroll
    for (int j = 0; j < 4; ++j) acc[i][j] = {};

  for (int k = 0; k < Kdim; k += 32) {
    v16bf a0 = frag16(pa0 + k, 8);
    v16bf a1 = frag16(pa1 + k, 8);
    v16bf b0 = frag16(pb0 + k, 8);
    v16bf b1 = frag16(pb1 + k, 8);
    v16bf b2 = frag16(pb2 + k, 8);
    v16bf b3 = frag16(pb3 + k, 8);
    acc[0][0] = __builtin_amdgcn_wmma_f32_16x16x32_bf16(false, a0, false, b0, (short)0, acc[0][0], false, false);
    acc[0][1] = __builtin_amdgcn_wmma_f32_16x16x32_bf16(false, a0, false, b1, (short)0, acc[0][1], false, false);
    acc[0][2] = __builtin_amdgcn_wmma_f32_16x16x32_bf16(false, a0, false, b2, (short)0, acc[0][2], false, false);
    acc[0][3] = __builtin_amdgcn_wmma_f32_16x16x32_bf16(false, a0, false, b3, (short)0, acc[0][3], false, false);
    acc[1][0] = __builtin_amdgcn_wmma_f32_16x16x32_bf16(false, a1, false, b0, (short)0, acc[1][0], false, false);
    acc[1][1] = __builtin_amdgcn_wmma_f32_16x16x32_bf16(false, a1, false, b1, (short)0, acc[1][1], false, false);
    acc[1][2] = __builtin_amdgcn_wmma_f32_16x16x32_bf16(false, a1, false, b2, (short)0, acc[1][2], false, false);
    acc[1][3] = __builtin_amdgcn_wmma_f32_16x16x32_bf16(false, a1, false, b3, (short)0, acc[1][3], false, false);
  }

  // C/D layout: VGPR r -> (lane<16 ? M=r : M=r+8), N = lane&15
  const int rofs = (lane >> 4) << 3;
  const int cofs = lane & 15;
#pragma unroll
  for (int i = 0; i < 2; ++i) {
#pragma unroll
    for (int j = 0; j < 4; ++j) {
      const int tc    = colBase + j * 16 + cofs;
      const int rbase = rowBase + i * 16 + rofs;
      const float bv  = bias ? bias[tc] : 0.0f;
      float vals[8];
#pragma unroll
      for (int r = 0; r < 8; ++r) {
        float v = acc[i][j][r] * alpha + bv;
        if (addsrc) v += addsrc[(long)(rbase + r) * ldadd + tc];
        if (relu)   v = fmaxf(v, 0.0f);
        vals[r] = v;
      }
      if (Cf) {
#pragma unroll
        for (int r = 0; r < 8; ++r)
          Cf[(long)z * zsC + (long)(rbase + r) * ldc + tc] = vals[r];
      }
      if (Cb) {
#pragma unroll
        for (int r = 0; r < 8; ++r)
          Cb[(long)z * zsCb + (long)(rbase + r) * ldcb + tc] = f32_to_bf16(vals[r]);
      }
      if (CbT) {  // 8 consecutive rows -> one packed 16B store at [tc][rbase]
        uint4 pk;
        pk.x = (unsigned)f32_to_bf16(vals[0]) | ((unsigned)f32_to_bf16(vals[1]) << 16);
        pk.y = (unsigned)f32_to_bf16(vals[2]) | ((unsigned)f32_to_bf16(vals[3]) << 16);
        pk.z = (unsigned)f32_to_bf16(vals[4]) | ((unsigned)f32_to_bf16(vals[5]) << 16);
        pk.w = (unsigned)f32_to_bf16(vals[6]) | ((unsigned)f32_to_bf16(vals[7]) << 16);
        *(uint4*)(CbT + (long)z * zsCT + (long)tc * ldcT + rbase) = pk;
      }
    }
  }
}

// f32 [R,C] -> bf16 [C,R] transpose-convert (R, C multiples of 32).
__global__ __launch_bounds__(256)
void transpose_cvt_kernel(const float* __restrict__ in, unsigned short* __restrict__ out,
                          int R, int C) {
  __shared__ unsigned short tile[32][34];
  const int c0 = blockIdx.x * 32, r0 = blockIdx.y * 32;
  const int tx = threadIdx.x & 31, ty = threadIdx.x >> 5;   // 32 x 8
  for (int i = ty; i < 32; i += 8)
    tile[i][tx] = f32_to_bf16(in[(long)(r0 + i) * C + c0 + tx]);
  __syncthreads();
  for (int i = ty; i < 32; i += 8)
    out[(long)(c0 + i) * R + r0 + tx] = tile[tx][i];
}

__global__ void cvt_f32_bf16_kernel(const float* __restrict__ in,
                                    unsigned short* __restrict__ out, long n) {
  long i = (long)blockIdx.x * blockDim.x + threadIdx.x;
  if (i < n) out[i] = f32_to_bf16(in[i]);
}

// Masked softmax over one (row, head) pair of 1024 scores -> bf16 probs.
__global__ __launch_bounds__(256)
void softmax_mask_kernel(const float* __restrict__ scores, const int* __restrict__ mask,
                         unsigned short* __restrict__ out, int ncols) {
  const int rid = blockIdx.x;           // l*H + h
  const int l = rid / HH, h = rid % HH;
  const float*    s  = scores + (long)l * HH * ncols + (long)h * ncols;
  const int*      mk = mask   + (long)l * ncols;
  unsigned short* o  = out    + (long)l * HH * ncols + (long)h * ncols;
  __shared__ float red[256];
  const int tid = threadIdx.x;

  float mx = -3.0e38f;
  for (int i = tid; i < ncols; i += 256) {
    float v = (mk[i] == 0) ? -1.0e9f : s[i];
    mx = fmaxf(mx, v);
  }
  red[tid] = mx; __syncthreads();
  for (int st = 128; st > 0; st >>= 1) {
    if (tid < st) red[tid] = fmaxf(red[tid], red[tid + st]);
    __syncthreads();
  }
  mx = red[0]; __syncthreads();

  float sum = 0.0f;
  for (int i = tid; i < ncols; i += 256) {
    float v = (mk[i] == 0) ? -1.0e9f : s[i];
    sum += __expf(v - mx);
  }
  red[tid] = sum; __syncthreads();
  for (int st = 128; st > 0; st >>= 1) {
    if (tid < st) red[tid] += red[tid + st];
    __syncthreads();
  }
  float inv = 1.0f / red[0];

  for (int i = tid; i < ncols; i += 256) {
    float v = (mk[i] == 0) ? -1.0e9f : s[i];
    o[i] = f32_to_bf16(__expf(v - mx) * inv);
  }
}

// LayerNorm over rows of width 256 (one block per row, one element per thread).
__global__ __launch_bounds__(256)
void layernorm_kernel(const float* __restrict__ x, const float* __restrict__ g,
                      const float* __restrict__ b, float* __restrict__ out, int ncols) {
  const int row = blockIdx.x, tid = threadIdx.x;
  __shared__ float red[256];
  float v = x[(long)row * ncols + tid];
  red[tid] = v; __syncthreads();
  for (int st = 128; st > 0; st >>= 1) {
    if (tid < st) red[tid] += red[tid + st];
    __syncthreads();
  }
  float mu = red[0] / (float)ncols; __syncthreads();
  float d = v - mu;
  red[tid] = d * d; __syncthreads();
  for (int st = 128; st > 0; st >>= 1) {
    if (tid < st) red[tid] += red[tid + st];
    __syncthreads();
  }
  float var = red[0] / (float)ncols;
  out[(long)row * ncols + tid] = d * rsqrtf(var + 1e-5f) * g[tid] + b[tid];
}

// ---------------------------------------------------------------------------
// Host-side helpers
// ---------------------------------------------------------------------------
static void gemm(hipStream_t s,
                 const void* A, long ldA, long zsA,
                 const void* BT, long ldBT, long zsB,
                 float* Cf, long ldc, long zsC,
                 unsigned short* Cb, long ldcb, long zsCb,
                 unsigned short* CbT, long ldcT, long zsCT,
                 const float* bias, const float* addsrc, long ldadd,
                 int M, int N, int K, float alpha, int relu, int batch) {
  dim3 grid((N + 127) / 128, M / 128, batch);
  gemm_bf16_kernel<<<grid, dim3(256), 0, s>>>(
      (const unsigned short*)A, ldA, zsA, (const unsigned short*)BT, ldBT, zsB,
      Cf, ldc, zsC, Cb, ldcb, zsCb, CbT, ldcT, zsCT,
      bias, addsrc, ldadd, N, K, alpha, relu);
}

static void tcvt(hipStream_t s, const float* in, unsigned short* outT, int R, int C) {
  transpose_cvt_kernel<<<dim3(C / 32, R / 32), dim3(256), 0, s>>>(in, outT, R, C);
}

static void cvt(hipStream_t s, const float* in, unsigned short* out, long n) {
  cvt_f32_bf16_kernel<<<dim3((unsigned)((n + 255) / 256)), dim3(256), 0, s>>>(in, out, n);
}

extern "C" void kernel_launch(void* const* d_in, const int* in_sizes, int n_in,
                              void* d_out, int out_size, void* d_ws, size_t ws_size,
                              hipStream_t stream) {
  (void)in_sizes; (void)n_in; (void)out_size;

  const float* lemb  = (const float*)d_in[0];
  const float* memb  = (const float*)d_in[1];
  const float* pemb  = (const float*)d_in[2];
  const int*   maskL = (const int*)d_in[3];
  const int*   maskP = (const int*)d_in[4];
  // 0=Wl 1=Wm 2=Wp 3=Wql 4=Wkl 5=Wvl 6=Wqp 7=Wkp 8=Wvp
  const float* W[9]; const float* bv[9];
  for (int i = 0; i < 9; ++i) { W[i] = (const float*)d_in[5 + 2 * i]; bv[i] = (const float*)d_in[6 + 2 * i]; }
  const float* Wml = (const float*)d_in[23]; const float* bml = (const float*)d_in[24];
  const float* Wmp = (const float*)d_in[25]; const float* bmp = (const float*)d_in[26];
  const float* g1  = (const float*)d_in[27]; const float* be1 = (const float*)d_in[28];
  const float* g2  = (const float*)d_in[29]; const float* be2 = (const float*)d_in[30];

  // ---- workspace arena ----
  char* ws = (char*)d_ws;
  size_t off = 0;
  auto carve = [&](size_t bytes) -> char* {
    char* p = ws + off;
    off += (bytes + 255) & ~(size_t)255;
    return p;
  };
  unsigned short* WbfT[9];                               // W^T [512][512]
  for (int i = 0; i < 9; ++i) WbfT[i] = (unsigned short*)carve((size_t)HIDDEN * HIDDEN * 2);
  unsigned short* WmlT = (unsigned short*)carve((size_t)DOUT * 2 * HIDDEN * 2);  // [256][1024]
  unsigned short* WmpT = (unsigned short*)carve((size_t)DOUT * 2 * HIDDEN * 2);
  unsigned short* lemb_bf = (unsigned short*)carve((size_t)LROWS * HIDDEN * 2);
  unsigned short* memb_bf = (unsigned short*)carve((size_t)MROWS * HIDDEN * 2);
  unsigned short* pemb_bf = (unsigned short*)carve((size_t)PROWS * HIDDEN * 2);
  unsigned short* lt_bf = (unsigned short*)carve((size_t)LROWS * HIDDEN * 2);
  unsigned short* mt_bf = (unsigned short*)carve((size_t)MROWS * HIDDEN * 2);
  unsigned short* pt_bf = (unsigned short*)carve((size_t)PROWS * HIDDEN * 2);
  unsigned short* Ql_bf = (unsigned short*)carve((size_t)LROWS * HIDDEN * 2);
  unsigned short* Kl_bf = (unsigned short*)carve((size_t)MROWS * HIDDEN * 2);
  unsigned short* Qp_bf = (unsigned short*)carve((size_t)PROWS * HIDDEN * 2);
  unsigned short* Kp_bf = (unsigned short*)carve((size_t)MROWS * HIDDEN * 2);
  unsigned short* VlT_bf = (unsigned short*)carve((size_t)HIDDEN * MROWS * 2);   // [512][1024]
  unsigned short* VpT_bf = (unsigned short*)carve((size_t)HIDDEN * MROWS * 2);
  float* scores = (float*)carve((size_t)LROWS * AWCOLS * 4);     // reused L then P side
  unsigned short* awl_bf = (unsigned short*)carve((size_t)LROWS * AWCOLS * 2);
  unsigned short* awp_bf = (unsigned short*)carve((size_t)PROWS * AWCOLS * 2);
  float* ctx_l = (float*)carve((size_t)LROWS * HIDDEN * 4);
  float* ctx_p = (float*)carve((size_t)PROWS * HIDDEN * 4);
  unsigned short* ctxlT_bf = (unsigned short*)carve((size_t)HIDDEN * LROWS * 2); // [512][4096]
  unsigned short* ctxpT_bf = (unsigned short*)carve((size_t)HIDDEN * PROWS * 2);
  unsigned short* mut_bf  = (unsigned short*)carve((size_t)LROWS * PROWS * 2);   // [L][P]
  unsigned short* mutT_bf = (unsigned short*)carve((size_t)PROWS * LROWS * 2);   // [P][L]
  unsigned short* finl_bf = (unsigned short*)carve((size_t)LROWS * HIDDEN * 2);
  unsigned short* finp_bf = (unsigned short*)carve((size_t)PROWS * HIDDEN * 2);
  float* pre = (float*)carve((size_t)LROWS * DOUT * 4);          // reused for both outputs
  if (off > ws_size) return;

  const long S = HIDDEN;      // 512
  const float scale = 0.125f; // 1/sqrt(64) == 1/H

  // 1) conversions: embeddings row-major bf16; all weights transposed bf16
  cvt(stream, lemb, lemb_bf, (long)LROWS * S);
  cvt(stream, memb, memb_bf, (long)MROWS * S);
  cvt(stream, pemb, pemb_bf, (long)PROWS * S);
  for (int i = 0; i < 9; ++i) tcvt(stream, W[i], WbfT[i], HIDDEN, HIDDEN);
  tcvt(stream, Wml, WmlT, 2 * HIDDEN, DOUT);
  tcvt(stream, Wmp, WmpT, 2 * HIDDEN, DOUT);

  // 2) input projections -> bf16 directly
  gemm(stream, lemb_bf, S, 0, WbfT[0], S, 0, nullptr,0,0, lt_bf, S,0, nullptr,0,0, bv[0], nullptr,0, LROWS, S, S, 1.0f, 0, 1);
  gemm(stream, memb_bf, S, 0, WbfT[1], S, 0, nullptr,0,0, mt_bf, S,0, nullptr,0,0, bv[1], nullptr,0, MROWS, S, S, 1.0f, 0, 1);
  gemm(stream, pemb_bf, S, 0, WbfT[2], S, 0, nullptr,0,0, pt_bf, S,0, nullptr,0,0, bv[2], nullptr,0, PROWS, S, S, 1.0f, 0, 1);

  // 3) Q/K row-major bf16; V transposed bf16 (B operand of ctx GEMM)
  gemm(stream, lt_bf, S, 0, WbfT[3], S, 0, nullptr,0,0, Ql_bf, S,0, nullptr,0,0, bv[3], nullptr,0, LROWS, S, S, 1.0f, 0, 1);
  gemm(stream, mt_bf, S, 0, WbfT[4], S, 0, nullptr,0,0, Kl_bf, S,0, nullptr,0,0, bv[4], nullptr,0, MROWS, S, S, 1.0f, 0, 1);
  gemm(stream, mt_bf, S, 0, WbfT[5], S, 0, nullptr,0,0, nullptr,0,0, VlT_bf, MROWS,0, bv[5], nullptr,0, MROWS, S, S, 1.0f, 0, 1);
  gemm(stream, pt_bf, S, 0, WbfT[6], S, 0, nullptr,0,0, Qp_bf, S,0, nullptr,0,0, bv[6], nullptr,0, PROWS, S, S, 1.0f, 0, 1);
  gemm(stream, mt_bf, S, 0, WbfT[7], S, 0, nullptr,0,0, Kp_bf, S,0, nullptr,0,0, bv[7], nullptr,0, MROWS, S, S, 1.0f, 0, 1);
  gemm(stream, mt_bf, S, 0, WbfT[8], S, 0, nullptr,0,0, nullptr,0,0, VpT_bf, MROWS,0, bv[8], nullptr,0, MROWS, S, S, 1.0f, 0, 1);

  // 4) scores L-side: per-head Q·K^T (B^T == K row-major) -> [L, H*M] f32
  gemm(stream, Ql_bf, S, DHEAD, Kl_bf, S, DHEAD,
       scores, AWCOLS, MROWS, nullptr,0,0, nullptr,0,0, nullptr, nullptr,0,
       LROWS, MROWS, DHEAD, scale, 0, HH);
  softmax_mask_kernel<<<dim3(LROWS * HH), dim3(256), 0, stream>>>(scores, maskL, awl_bf, MROWS);

  // 5) scores P-side (reuse scratch)
  gemm(stream, Qp_bf, S, DHEAD, Kp_bf, S, DHEAD,
       scores, AWCOLS, MROWS, nullptr,0,0, nullptr,0,0, nullptr, nullptr,0,
       PROWS, MROWS, DHEAD, scale, 0, HH);
  softmax_mask_kernel<<<dim3(PROWS * HH), dim3(256), 0, stream>>>(scores, maskP, awp_bf, MROWS);

  // 6) context: aw·V with ReLU -> f32 ctx (fin residual) + transposed bf16 ctx^T
  gemm(stream, awl_bf, AWCOLS, MROWS, VlT_bf, MROWS, (long)DHEAD * MROWS,
       ctx_l, S, DHEAD, nullptr,0,0, ctxlT_bf, LROWS, (long)DHEAD * LROWS,
       nullptr, nullptr,0, LROWS, DHEAD, MROWS, 1.0f, 1, HH);
  gemm(stream, awp_bf, AWCOLS, MROWS, VpT_bf, MROWS, (long)DHEAD * MROWS,
       ctx_p, S, DHEAD, nullptr,0,0, ctxpT_bf, PROWS, (long)DHEAD * PROWS,
       nullptr, nullptr,0, PROWS, DHEAD, MROWS, 1.0f, 1, HH);

  // 7) mutual = (1/H) * awl · awp^T : K = 8192, bf16 out both orders
  gemm(stream, awl_bf, AWCOLS, 0, awp_bf, AWCOLS, 0,
       nullptr,0,0, mut_bf, PROWS,0, mutT_bf, LROWS,0,
       nullptr, nullptr,0, LROWS, PROWS, AWCOLS, 0.125f, 0, 1);

  // 8) fin_l = ctx_l + mutual·ctx_p ; fin_p = ctx_p + mutual^T·ctx_l  (bf16 out)
  gemm(stream, mut_bf, PROWS, 0, ctxpT_bf, PROWS, 0,
       nullptr,0,0, finl_bf, S,0, nullptr,0,0, nullptr, ctx_l, S,
       LROWS, S, PROWS, 1.0f, 0, 1);
  gemm(stream, mutT_bf, LROWS, 0, ctxlT_bf, LROWS, 0,
       nullptr,0,0, finp_bf, S,0, nullptr,0,0, nullptr, ctx_p, S,
       PROWS, S, LROWS, 1.0f, 0, 1);

  float* out_l = (float*)d_out;
  float* out_p = out_l + (long)LROWS * DOUT;

  // 9) out_l = LN( [lt | fin_l] @ Wml + bml ), concat split over K
  gemm(stream, lt_bf, S, 0, WmlT, 2 * S, 0,
       pre, DOUT, 0, nullptr,0,0, nullptr,0,0, nullptr, nullptr,0, LROWS, DOUT, S, 1.0f, 0, 1);
  gemm(stream, finl_bf, S, 0, WmlT + HIDDEN, 2 * S, 0,
       pre, DOUT, 0, nullptr,0,0, nullptr,0,0, bml, pre, DOUT, LROWS, DOUT, S, 1.0f, 0, 1);
  layernorm_kernel<<<dim3(LROWS), dim3(256), 0, stream>>>(pre, g1, be1, out_l, DOUT);

  // 10) out_p = LN( [fin_p | pt] @ Wmp + bmp )
  gemm(stream, finp_bf, S, 0, WmpT, 2 * S, 0,
       pre, DOUT, 0, nullptr,0,0, nullptr,0,0, nullptr, nullptr,0, PROWS, DOUT, S, 1.0f, 0, 1);
  gemm(stream, pt_bf, S, 0, WmpT + HIDDEN, 2 * S, 0,
       pre, DOUT, 0, nullptr,0,0, nullptr,0,0, bmp, pre, DOUT, PROWS, DOUT, S, 1.0f, 0, 1);
  layernorm_kernel<<<dim3(PROWS), dim3(256), 0, stream>>>(pre, g2, be2, out_p, DOUT);
}
